// JointDistributionGenerator_21122649162222
// MI455X (gfx1250) — matrix operand order repaired and checked by reading, not compiled, not run on hardware
//
#include <hip/hip_runtime.h>
#include <hip/hip_bf16.h>

typedef __attribute__((ext_vector_type(16))) __bf16   v16bf;
typedef __attribute__((ext_vector_type(8)))  float    v8f;
typedef __attribute__((ext_vector_type(8)))  unsigned v8u;

#define LOG2PI_F 1.8378770664093453f

// ---------- helpers ----------
__device__ __forceinline__ unsigned short f2bf_bits(float f) {
    unsigned u = __float_as_uint(f);
    unsigned r = (u + 0x7fffu + ((u >> 16) & 1u)) >> 16;   // round-to-nearest-even
    return (unsigned short)r;
}
__device__ __forceinline__ __bf16 f2bf(float f) {
    unsigned short s = f2bf_bits(f);
    __bf16 b;
    __builtin_memcpy(&b, &s, 2);
    return b;
}
__device__ __forceinline__ unsigned hashu(unsigned x) {
    x ^= x >> 16; x *= 0x7feb352dU;
    x ^= x >> 15; x *= 0x846ca68bU;
    x ^= x >> 16; return x;
}
__device__ __forceinline__ float u01(unsigned h) {
    return (float)(h >> 8) * 5.9604645e-08f + 2.9802322e-08f; // (0,1)
}
// A-operand element e of a 16-bit 16x32 WMMA A matrix maps to K = e + (e&8) + 8*half
__device__ __forceinline__ int kof(int e, int half) { return e + (e & 8) + 8 * half; }

union BfPun { v16bf b; v8u u; };

// ---------- setup: per-component factorization, K=16 blocks x 32 lanes ----------
__device__ void chol32(float (*S)[33], float (*L)[33], int l) {
    for (int j = 0; j < 32; ++j) {
        float s = S[j][j];
        for (int p = 0; p < j; ++p) { float v = L[j][p]; s -= v * v; }
        float d = sqrtf(fmaxf(s, 1e-20f));
        if (l == j) L[j][j] = d;
        if (l > j) {
            float t = S[l][j];
            for (int p = 0; p < j; ++p) t -= L[l][p] * L[j][p];
            L[l][j] = t / d;
        }
        if (l < j) L[l][j] = 0.0f;
        __syncthreads();
    }
}

__global__ void gmm_setup_kernel(const float* __restrict__ cov,
                                 const float* __restrict__ means,
                                 unsigned short* __restrict__ bop,
                                 float* __restrict__ c1o,   // c1[k][i] = (L11inv @ mean1)[i]
                                 float* __restrict__ dof,   // d[k][i]  = mean1[i] - (A @ mean2)[i]
                                 float* __restrict__ logdet) {
    const int k = blockIdx.x;      // component
    const int l = threadIdx.x;     // 0..31
    __shared__ float C22[32][33], C12[32][33], C11[32][33];
    __shared__ float Am[32][33], L[32][33], Li[32][33], CC[32][33], Lc[32][33];
    __shared__ float mv1[32], mv2[32];

    const float* ck = cov + (size_t)k * 64 * 64;
    for (int i = 0; i < 32; ++i) {
        C22[i][l] = ck[i * 64 + l];              // sig x sig
        C12[i][l] = ck[(32 + i) * 64 + l];       // comp x sig
        C11[i][l] = ck[(32 + i) * 64 + 32 + l];  // comp x comp
    }
    mv1[l] = means[k * 64 + 32 + l];             // mean over comp dims
    mv2[l] = means[k * 64 + l];                  // mean over sig dims
    __syncthreads();

    chol32(C22, L, l);           // L = chol(cov22)
    __syncthreads();

    // A[l][:] = solve(cov22, cov12[l][:])  (lane l owns row l)
    {
        float y[32], x[32];
        for (int r = 0; r < 32; ++r) {
            float t = C12[l][r];
            for (int p = 0; p < r; ++p) t -= L[r][p] * y[p];
            y[r] = t / L[r][r];
        }
        for (int r = 31; r >= 0; --r) {
            float t = y[r];
            for (int p = r + 1; p < 32; ++p) t -= L[p][r] * x[p];
            x[r] = t / L[r][r];
        }
        for (int r = 0; r < 32; ++r) Am[l][r] = x[r];
    }
    __syncthreads();

    // cond_cov = cov11 - A * cov12^T
    for (int c = 0; c < 32; ++c) {
        float s = C11[l][c];
        for (int j = 0; j < 32; ++j) s -= Am[l][j] * C12[c][j];
        CC[l][c] = s;
    }
    __syncthreads();

    chol32(CC, Lc, l);           // Lc = chol(cond_cov)
    __syncthreads();
    chol32(C11, L, l);           // L = chol(cov11) == L11 now
    __syncthreads();

    // Li = inv(L11) (lower). Lane l owns column l.
    {
        float x[32];
        for (int r = 0; r < 32; ++r) x[r] = 0.0f;
        x[l] = 1.0f / L[l][l];
        for (int r = l + 1; r < 32; ++r) {
            float t = 0.0f;
            for (int p = l; p < r; ++p) t -= L[r][p] * x[p];
            x[r] = t / L[r][r];
        }
        for (int r = 0; r < 32; ++r) Li[r][l] = (r >= l) ? x[r] : 0.0f;
    }
    __syncthreads();

    if (l == 0) {
        float s = 0.0f;
        for (int i = 0; i < 32; ++i) s += __logf(L[i][i]);
        logdet[k] = s;
    }
    // fold means into linear constants (row l)
    {
        float s = 0.0f;
        for (int j = 0; j <= l; ++j) s += Li[l][j] * mv1[j];
        c1o[k * 32 + l] = s;                       // z = L11inv*x - c1
        float s2 = mv1[l];
        for (int j = 0; j < 32; ++j) s2 -= Am[l][j] * mv2[j];
        dof[k * 32 + l] = s2;                      // x_out = A*x + d + Lc*eps
    }

    // Emit bf16 B-operands in WMMA lane layout:
    // matid 0 = A, 1 = L11inv, 2 = Lc.  For out_i = sum_j M[i][j] in_j, B[K=j][N=i] = M[N][K].
    const int khalf = l >> 4, nn = l & 15;
    for (int matid = 0; matid < 3; ++matid) {
        float (*M)[33] = (matid == 0) ? Am : (matid == 1) ? Li : Lc;
        for (int ch = 0; ch < 2; ++ch) {
            const int N = nn + 16 * ch;
            const size_t base = ((size_t)(matid * 16 + k) * 2 + ch) * 512 + (size_t)l * 16;
            for (int e = 0; e < 16; ++e) {
                const int K = kof(e, khalf);
                bop[base + e] = f2bf_bits(M[N][K]);
            }
        }
    }
}

// ---------- main: one wave per 16-row tile ----------
__global__ void gmm_sample_kernel(const float* __restrict__ current,
                                  const float* __restrict__ weights,
                                  const unsigned char* __restrict__ blob, // ws: bop|c1|d
                                  const float* __restrict__ logdet,
                                  float* __restrict__ out) {
    extern __shared__ unsigned char smem[];
    unsigned short* sbop = (unsigned short*)smem;           // 98304 B
    float* sc1 = (float*)(smem + 98304);                    // 2048 B
    float* sd  = (float*)(smem + 100352);                   // 2048 B
    int*   ssel = (int*)(smem + 102400);                    // 4 waves x 16 rows

    // ---- async stage of operand tables into LDS (6400 x b128, ASYNCcnt path) ----
    {
        const unsigned lbase = (unsigned)(uintptr_t)smem;   // LDS aperture: addr[31:0] = LDS offset
        for (int i = threadIdx.x; i < 6400; i += blockDim.x) {
            const unsigned ldsoff = lbase + (unsigned)i * 16u;
            const unsigned voff   = (unsigned)i * 16u;
            asm volatile("global_load_async_to_lds_b128 %0, %1, %2 offset:0"
                         :: "v"(ldsoff), "v"(voff), "s"(blob) : "memory");
        }
        asm volatile("s_wait_asynccnt 0x0" ::: "memory");
    }
    __syncthreads();

    const int lane = threadIdx.x & 31;
    const int wv   = threadIdx.x >> 5;
    const int half = lane >> 4;
    const int n    = lane & 15;
    const int tile = blockIdx.x * 4 + wv;
    const int r0   = tile * 16;
    const int mrow = r0 + n;                 // batch row carried by this lane (A-operand)

    __builtin_prefetch(&current[(size_t)mrow * 64], 0, 3);

    // load x_sig tile in WMMA A layout: two contiguous 8-dword runs -> 4 x b128
    float xa[16];
    {
        const float4* rp = (const float4*)(current + (size_t)mrow * 64);
        const float4 q0 = rp[2 * half], q1 = rp[2 * half + 1];
        const float4 q2 = rp[4 + 2 * half], q3 = rp[5 + 2 * half];
        xa[0] = q0.x; xa[1] = q0.y; xa[2]  = q0.z; xa[3]  = q0.w;
        xa[4] = q1.x; xa[5] = q1.y; xa[6]  = q1.z; xa[7]  = q1.w;
        xa[8] = q2.x; xa[9] = q2.y; xa[10] = q2.z; xa[11] = q2.w;
        xa[12] = q3.x; xa[13] = q3.y; xa[14] = q3.z; xa[15] = q3.w;
    }
    BfPun X;
#pragma unroll
    for (int e = 0; e < 16; ++e) X.b[e] = f2bf(xa[e]);

    // ---------------- phase 1: logprob + Gumbel argmax over k ----------------
    float best[8];
    int   bidx[8];
#pragma unroll
    for (int r = 0; r < 8; ++r) { best[r] = -3.0e38f; bidx[r] = 0; }

    for (int k = 0; k < 16; ++k) {
        const v16bf b0 = *(const v16bf*)(sbop + ((size_t)(16 + k) * 2 + 0) * 512 + (size_t)lane * 16);
        const v16bf b1 = *(const v16bf*)(sbop + ((size_t)(16 + k) * 2 + 1) * 512 + (size_t)lane * 16);

        const float cA = -sc1[k * 32 + n];
        const float cB = -sc1[k * 32 + 16 + n];
        const v8f c0 = {cA, cA, cA, cA, cA, cA, cA, cA};
        const v8f c1v = {cB, cB, cB, cB, cB, cB, cB, cB};

        v8f d0 = __builtin_amdgcn_wmma_f32_16x16x32_bf16(false, X.b, false, b0, (short)0, c0, false, false);
        v8f d1 = __builtin_amdgcn_wmma_f32_16x16x32_bf16(false, X.b, false, b1, (short)0, c1v, false, false);

        const float ldk = logdet[k];
        const float wk  = weights[k];
#pragma unroll
        for (int r = 0; r < 8; ++r) {
            float t = d0[r] * d0[r] + d1[r] * d1[r];
            t += __shfl_xor(t, 1, 32);
            t += __shfl_xor(t, 2, 32);
            t += __shfl_xor(t, 4, 32);
            t += __shfl_xor(t, 8, 32);
            const float lp = -0.5f * t - ldk - 16.0f * LOG2PI_F;
            const float pi = wk * __expf(lp);                      // reference feeds this as logits
            const unsigned row = (unsigned)(r0 + r + 8 * half);
            const float g  = -__logf(-__logf(u01(hashu(row * 0x9E3779B9u ^ (unsigned)(k * 0x85ebca6b) ^ 0xC001F00Du))));
            const float sc = pi + g;
            if (sc > best[r]) { best[r] = sc; bidx[r] = k; }
        }
    }

    // exchange selection to lane-indexed layout
    if (n == 0) {
#pragma unroll
        for (int r = 0; r < 8; ++r) ssel[wv * 16 + r + 8 * half] = bidx[r];
    }
    __syncthreads();
    const int sel_l = ssel[wv * 16 + n];     // chosen component for row mrow

    // gaussian eps for this row, in A-operand layout (consistent across both lane halves)
    BfPun E;
#pragma unroll
    for (int e = 0; e < 16; ++e) {
        const unsigned key = hashu((unsigned)(mrow * 64 + kof(e, half)) * 0x9E3779B9u + 0x1234567u);
        const float u1 = u01(key);
        const float u2 = u01(hashu(key ^ 0xBEEFCAFEu));
        E.b[e] = f2bf(sqrtf(-2.0f * __logf(u1)) * __cosf(6.28318530718f * u2));
    }

    // ---------------- phase 2: x = d[sel] + A[sel]*x + Lc[sel]*eps via masked WMMA ----------------
    v8f acc0 = {}, acc1 = {};
    for (int k = 0; k < 16; ++k) {
        const bool me = (sel_l == k);
        BfPun a2, ae;
#pragma unroll
        for (int i = 0; i < 8; ++i) {          // dword-granular masking (row selector is per-lane)
            a2.u[i] = me ? X.u[i] : 0u;
            ae.u[i] = me ? E.u[i] : 0u;
        }
        const v16bf bA0 = *(const v16bf*)(sbop + ((size_t)(k) * 2 + 0) * 512 + (size_t)lane * 16);
        const v16bf bA1 = *(const v16bf*)(sbop + ((size_t)(k) * 2 + 1) * 512 + (size_t)lane * 16);
        const v16bf bL0 = *(const v16bf*)(sbop + ((size_t)(32 + k) * 2 + 0) * 512 + (size_t)lane * 16);
        const v16bf bL1 = *(const v16bf*)(sbop + ((size_t)(32 + k) * 2 + 1) * 512 + (size_t)lane * 16);

        acc0 = __builtin_amdgcn_wmma_f32_16x16x32_bf16(false, a2.b, false, bA0, (short)0, acc0, false, false);
        acc1 = __builtin_amdgcn_wmma_f32_16x16x32_bf16(false, a2.b, false, bA1, (short)0, acc1, false, false);
        acc0 = __builtin_amdgcn_wmma_f32_16x16x32_bf16(false, ae.b, false, bL0, (short)0, acc0, false, false);
        acc1 = __builtin_amdgcn_wmma_f32_16x16x32_bf16(false, ae.b, false, bL1, (short)0, acc1, false, false);

#pragma unroll
        for (int r = 0; r < 8; ++r) {
            if (bidx[r] == k) {                 // add loc constant d[sel] (C/D row m = r + 8*half)
                acc0[r] += sd[k * 32 + n];
                acc1[r] += sd[k * 32 + 16 + n];
            }
        }
    }

    // ---------------- stores ----------------
#pragma unroll
    for (int r = 0; r < 8; ++r) {
        const int row = r0 + r + 8 * half;
        out[(size_t)row * 64 + 32 + n] = acc0[r];
        out[(size_t)row * 64 + 48 + n] = acc1[r];
    }
    {   // pass-through of sig dims, 4 x b128
        float4* op = (float4*)(out + (size_t)mrow * 64);
        float4 q;
        q.x = xa[0];  q.y = xa[1];  q.z = xa[2];  q.w = xa[3];  op[2 * half] = q;
        q.x = xa[4];  q.y = xa[5];  q.z = xa[6];  q.w = xa[7];  op[2 * half + 1] = q;
        q.x = xa[8];  q.y = xa[9];  q.z = xa[10]; q.w = xa[11]; op[4 + 2 * half] = q;
        q.x = xa[12]; q.y = xa[13]; q.z = xa[14]; q.w = xa[15]; op[5 + 2 * half] = q;
    }
}

// ---------- host ----------
extern "C" void kernel_launch(void* const* d_in, const int* in_sizes, int n_in,
                              void* d_out, int out_size, void* d_ws, size_t ws_size,
                              hipStream_t stream) {
    const float* current = (const float*)d_in[1];
    const float* means   = (const float*)d_in[3];
    const float* cov     = (const float*)d_in[4];
    const float* weights = (const float*)d_in[5];
    float* out = (float*)d_out;

    unsigned short* bop = (unsigned short*)d_ws;                 // 98304 B: 3 mats x 16 k x 2 ch x 512 bf16
    float* c1 = (float*)((char*)d_ws + 98304);                   // 16 x 32 f32
    float* dv = (float*)((char*)d_ws + 100352);                  // 16 x 32 f32
    float* ld = (float*)((char*)d_ws + 102400);                  // 16 f32

    gmm_setup_kernel<<<16, 32, 0, stream>>>(cov, means, bop, c1, dv, ld);

    const int Bn     = in_sizes[1] / 64;       // 131072
    const int tiles  = Bn / 16;                // 8192
    const int blocks = tiles / 4;              // 2048 (4 waves per block)
    const size_t shmem = 102400 + 4 * 16 * sizeof(int);
    gmm_sample_kernel<<<blocks, 128, shmem, stream>>>(current, weights,
                                                      (const unsigned char*)d_ws, ld, out);
}
